// KNNAttention_10136122818777
// MI455X (gfx1250) — matrix operand order repaired and checked by reading, not compiled, not run on hardware
//
#include <hip/hip_runtime.h>
#include <hip/hip_bf16.h>
#include <math.h>

// ---------------------------------------------------------------------------
// KNN-augmented attention (memorizing-transformers style), MI455X / gfx1250.
// b=2 n=2048 dim=512 h=8 d=64 topk=32. HBM-bound on k_mem/v_mem (536 MB);
// all GEMM-shaped work runs on v_wmma_f32_16x16x32_f16 (wave32); local-attn
// K/V tiles are staged with GLOBAL_LOAD_ASYNC_TO_LDS_B128 double-buffering.
// ---------------------------------------------------------------------------

typedef __attribute__((ext_vector_type(16))) _Float16 v16h;
typedef __attribute__((ext_vector_type(8)))  float    v8f;

#define MASKVAL (-3.402823466e38f)

__device__ __forceinline__ v8f wmma_f16(v16h a, v16h b, v8f c) {
  // D = A(16x32) * B(32x16) + C, f32 accumulate
  return __builtin_amdgcn_wmma_f32_16x16x32_f16(
      false, a, false, b, (short)0, c, false, false);
}

// Async global->LDS copy, 16B per lane, tracked by ASYNCcnt (ISA 10. / 15.18.3)
__device__ __forceinline__ void async_load_b128(unsigned lds_addr,
                                                const void* gaddr) {
  asm volatile("global_load_async_to_lds_b128 %0, %1, off"
               :
               : "v"(lds_addr), "v"(gaddr)
               : "memory");
}
__device__ __forceinline__ void wait_async0() {
  asm volatile("s_wait_asynccnt 0x0" ::: "memory");
}
// Generic-pointer low 32 bits == wave LDS byte offset (aperture rule).
__device__ __forceinline__ unsigned lds_off(const void* p) {
  return (unsigned)(uintptr_t)p;
}

// A-matrix 16x32 f16 per-lane K index for element e (0..15 of v16h).
// Lanes 0-15: K 0-15; lanes 16-31: K 16-31 (sequential, matches B).
__device__ __forceinline__ int a_kidx(int e, int lane) {
  return ((lane >> 4) << 4) + e;
}
// B-matrix 32x16 f16: lane&15 = column n; K sequential, halves split at K=16.
__device__ __forceinline__ int b_kidx(int e, int lane) {
  return ((lane >> 4) << 4) + e;
}

// A fragment from row-major [M][K] f16 buffer (contiguous: 2x ds_load_b128)
__device__ __forceinline__ v16h load_a_frag(const _Float16* base, int row0,
                                            int ld, int lane, int koff) {
  v16h f;
  int m = row0 + (lane & 15);
#pragma unroll
  for (int e = 0; e < 16; ++e) f[e] = base[m * ld + koff + a_kidx(e, lane)];
  return f;
}
// B fragment, B^T stored row-major [N][K] (contiguous: 2x ds_load_b128)
__device__ __forceinline__ v16h load_b_fragT(const _Float16* base, int n0,
                                             int ld, int lane, int koff) {
  v16h f;
  int n = n0 + (lane & 15);
#pragma unroll
  for (int e = 0; e < 16; ++e) f[e] = base[n * ld + koff + b_kidx(e, lane)];
  return f;
}
// B fragment, B stored row-major [K][N] (strided; used only for P@V V-tile)
__device__ __forceinline__ v16h load_b_frag(const _Float16* base, int n0,
                                            int ld, int lane, int koff) {
  v16h f;
  int n = n0 + (lane & 15);
#pragma unroll
  for (int e = 0; e < 16; ++e) f[e] = base[(koff + b_kidx(e, lane)) * ld + n];
  return f;
}

// Stage a 32(K) x 128(N) fp32 weight tile into LDS transposed [N][K] f16.
// 4x4 micro-blocks: 4 coalesced float4 reads -> 4x 8-byte LDS stores.
__device__ __forceinline__ void stage_wtile_T(const float* src, int ldw,
                                              _Float16* Bst, int tid) {
  int c = (tid & 31) * 4;   // n offset 0..124
  int r = (tid >> 5) * 4;   // k offset 0..28
  float uu[4][4];
#pragma unroll
  for (int i = 0; i < 4; ++i)
    *reinterpret_cast<float4*>(uu[i]) =
        *reinterpret_cast<const float4*>(src + (size_t)(r + i) * ldw + c);
#pragma unroll
  for (int j = 0; j < 4; ++j) {
    _Float16 tmp[4];
#pragma unroll
    for (int i = 0; i < 4; ++i) tmp[i] = (_Float16)uu[i][j];
    *reinterpret_cast<float2*>(&Bst[(c + j) * 32 + r]) =
        *reinterpret_cast<const float2*>(tmp);
  }
}

// ---------------------------------------------------------------------------
// K1: QKV projection. C(4096x1536) = x(4096x512) @ [Wq | Wkv], f16 out per head.
// grid(12,64) block 256 (8 waves, 2x4 sub-tiles of 32x32). Tile 64(M)x128(N).
// ---------------------------------------------------------------------------
__global__ __launch_bounds__(256) void k_qkv_gemm(
    const float* __restrict__ x, const float* __restrict__ Wq,
    const float* __restrict__ Wkv, _Float16* __restrict__ qf,
    _Float16* __restrict__ kf, _Float16* __restrict__ vf) {
  __shared__ _Float16 As[64 * 32];
  __shared__ _Float16 Bst[128 * 32];  // B transposed [N][K]
  const int tid = threadIdx.x, lane = tid & 31, wave = tid >> 5;
  const int m0 = blockIdx.y * 64;
  const int n0 = blockIdx.x * 128;
  const int wm = wave >> 2, wn = wave & 3;

  const float* Wsrc;
  int ldw, c0;
  if (n0 < 512) { Wsrc = Wq;  ldw = 512;  c0 = n0; }
  else          { Wsrc = Wkv; ldw = 1024; c0 = n0 - 512; }

  v8f acc[2][2] = {};
  for (int kk = 0; kk < 512; kk += 32) {
    __syncthreads();
    {  // stage A: 64x32 fp32 -> f16 row-major
      int r = tid >> 2, c = (tid & 3) * 8;
      const float* src = x + (size_t)(m0 + r) * 512 + kk + c;
      float4 u0 = *reinterpret_cast<const float4*>(src);
      float4 u1 = *reinterpret_cast<const float4*>(src + 4);
      _Float16* dst = &As[r * 32 + c];
      dst[0] = (_Float16)u0.x; dst[1] = (_Float16)u0.y;
      dst[2] = (_Float16)u0.z; dst[3] = (_Float16)u0.w;
      dst[4] = (_Float16)u1.x; dst[5] = (_Float16)u1.y;
      dst[6] = (_Float16)u1.z; dst[7] = (_Float16)u1.w;
    }
    stage_wtile_T(Wsrc + (size_t)kk * ldw + c0, ldw, Bst, tid);
    __syncthreads();
    v16h a0 = load_a_frag(As, wm * 32, 32, lane, 0);
    v16h a1 = load_a_frag(As, wm * 32 + 16, 32, lane, 0);
    v16h b0 = load_b_fragT(Bst, wn * 32, 32, lane, 0);
    v16h b1 = load_b_fragT(Bst, wn * 32 + 16, 32, lane, 0);
    acc[0][0] = wmma_f16(a0, b0, acc[0][0]);
    acc[0][1] = wmma_f16(a0, b1, acc[0][1]);
    acc[1][0] = wmma_f16(a1, b0, acc[1][0]);
    acc[1][1] = wmma_f16(a1, b1, acc[1][1]);
  }
  // epilogue: scatter into per-head [b][h][n][64] f16 q/k/v
#pragma unroll
  for (int i = 0; i < 2; ++i)
#pragma unroll
    for (int j = 0; j < 2; ++j)
#pragma unroll
      for (int r = 0; r < 8; ++r) {
        int ml = wm * 32 + i * 16 + r + ((lane >> 4) << 3);
        int nl = wn * 32 + j * 16 + (lane & 15);
        int M = m0 + ml, N = n0 + nl;
        int bb = M >> 11, pos = M & 2047;
        int which = N >> 9, cc = N & 511, hh = cc >> 6, dd = cc & 63;
        _Float16* dst = (which == 0) ? qf : (which == 1) ? kf : vf;
        dst[((size_t)(bb * 8 + hh) * 2048 + pos) * 64 + dd] =
            (_Float16)acc[i][j][r];
      }
}

// ---------------------------------------------------------------------------
// K2: local causal flash attention + T5 relative bias.
// grid(32 qblocks, 16 bh), block 128 (4 waves x 16 query rows). 32 keys/step.
// K/V tiles double-buffered via GLOBAL_LOAD_ASYNC_TO_LDS_B128.
// ---------------------------------------------------------------------------
__global__ __launch_bounds__(128) void k_local_attn(
    const _Float16* __restrict__ qf, const _Float16* __restrict__ kf,
    const _Float16* __restrict__ vf, const float* __restrict__ rel_emb,
    float* __restrict__ localv) {
  __shared__ float    biasT[2048];         // bias(i-j), pre-scaled by sqrt(d)
  __shared__ _Float16 Qs[64 * 64];
  __shared__ _Float16 Ks[2][32 * 64];      // ping-pong
  __shared__ _Float16 Vs[2][32 * 64];
  __shared__ _Float16 Pb[4][16 * 32];      // per-wave P transpose buffer

  const int tid = threadIdx.x, lane = tid & 31, wave = tid >> 5;
  const int qb = blockIdx.x;    // 0..31
  const int bh = blockIdx.y;    // 0..15
  const int h  = bh & 7;
  const size_t base = (size_t)bh * 2048 * 64;

  // T5 bucket bias table: n = i-j, scaled by sqrt(d)=8 (reference adds post-scale)
  for (int dlt = tid; dlt < 2048; dlt += 128) {
    int bucket;
    if (dlt < 16) bucket = dlt;
    else {
      float v = logf((float)dlt * (1.0f / 16.0f) + 1e-20f) *
                (16.0f / logf(8.0f));
      bucket = 16 + (int)v;
      if (bucket > 31) bucket = 31;
    }
    biasT[dlt] = rel_emb[bucket * 8 + h] * 8.0f;
  }
  {  // stage Q 64x64 f16
    int r = tid >> 1, c = (tid & 1) * 32;
    const float4* s = reinterpret_cast<const float4*>(
        qf + base + (size_t)(qb * 64 + r) * 64 + c);
    float4* d = reinterpret_cast<float4*>(&Qs[r * 64 + c]);
    d[0] = s[0]; d[1] = s[1]; d[2] = s[2]; d[3] = s[3];
  }

  // async stage of one 32x64 K tile + V tile into buffer p (32B per thread each)
  const int sr = tid >> 2, sc = (tid & 3) * 16;
  auto stage_kv = [&](int p, int kb) {
    size_t goff = base + (size_t)(kb * 32 + sr) * 64 + sc;
    async_load_b128(lds_off(&Ks[p][sr * 64 + sc]),     kf + goff);
    async_load_b128(lds_off(&Ks[p][sr * 64 + sc + 8]), kf + goff + 8);
    async_load_b128(lds_off(&Vs[p][sr * 64 + sc]),     vf + goff);
    async_load_b128(lds_off(&Vs[p][sr * 64 + sc + 8]), vf + goff + 8);
  };

  const int nkb = (qb + 1) * 2;  // 32-key blocks up to (and incl.) diagonal
  stage_kv(0, 0);
  wait_async0();
  __syncthreads();

  v16h qfr0 = load_a_frag(Qs, wave * 16, 64, lane, 0);
  v16h qfr1 = load_a_frag(Qs, wave * 16, 64, lane, 32);

  float mrow[8], lrow[8];
  v8f acc[4] = {};
#pragma unroll
  for (int r = 0; r < 8; ++r) { mrow[r] = MASKVAL; lrow[r] = 0.0f; }

  for (int kb = 0; kb < nkb; ++kb) {
    const int cur = kb & 1;
    if (kb + 1 < nkb) stage_kv(cur ^ 1, kb + 1);  // prefetch next tile (async)
    const _Float16* Kc = Ks[cur];
    const _Float16* Vc = Vs[cur];

    // sim tiles: keys [kb*32,+16) and [+16,+32), contraction over d=64
    v8f s0 = {}, s1 = {};
    s0 = wmma_f16(qfr0, load_b_fragT(Kc, 0, 64, lane, 0), s0);
    s0 = wmma_f16(qfr1, load_b_fragT(Kc, 0, 64, lane, 32), s0);
    s1 = wmma_f16(qfr0, load_b_fragT(Kc, 16, 64, lane, 0), s1);
    s1 = wmma_f16(qfr1, load_b_fragT(Kc, 16, 64, lane, 32), s1);

    const int qrow0 = qb * 64 + wave * 16 + ((lane >> 4) << 3);
    const int kp0 = kb * 32 + (lane & 15);
    float p0[8], p1[8], alpha[8];
#pragma unroll
    for (int r = 0; r < 8; ++r) {
      int qp = qrow0 + r;
      int d0 = qp - kp0, d1 = qp - (kp0 + 16);
      float v0 = (d0 >= 0) ? s0[r] * 0.125f + biasT[d0] : MASKVAL;
      float v1 = (d1 >= 0) ? s1[r] * 0.125f + biasT[d1] : MASKVAL;
      float mx = fmaxf(v0, v1);
#pragma unroll
      for (int s = 8; s >= 1; s >>= 1) mx = fmaxf(mx, __shfl_xor(mx, s, 32));
      float nm = fmaxf(mrow[r], mx);
      float a  = expf(mrow[r] - nm);
      float e0 = expf(v0 - nm), e1 = expf(v1 - nm);
      float rs = e0 + e1;
#pragma unroll
      for (int s = 8; s >= 1; s >>= 1) rs += __shfl_xor(rs, s, 32);
      mrow[r] = nm;
      lrow[r] = lrow[r] * a + rs;
      alpha[r] = a;
      p0[r] = e0; p1[r] = e1;
    }
#pragma unroll
    for (int c = 0; c < 4; ++c)
#pragma unroll
      for (int r = 0; r < 8; ++r) acc[c][r] *= alpha[r];

    // P (C-layout) -> LDS -> A-layout, then acc += P @ V  (K = 32 keys)
    _Float16* pb = Pb[wave];
#pragma unroll
    for (int r = 0; r < 8; ++r) {
      int rl = r + ((lane >> 4) << 3);
      pb[rl * 32 + (lane & 15)]      = (_Float16)p0[r];
      pb[rl * 32 + 16 + (lane & 15)] = (_Float16)p1[r];
    }
    v16h pa = load_a_frag(pb, 0, 32, lane, 0);
#pragma unroll
    for (int c = 0; c < 4; ++c)
      acc[c] = wmma_f16(pa, load_b_frag(Vc, c * 16, 64, lane, 0), acc[c]);

    wait_async0();     // next tile resident in other buffer
    __syncthreads();   // everyone done reading current buffer
  }

#pragma unroll
  for (int c = 0; c < 4; ++c)
#pragma unroll
    for (int r = 0; r < 8; ++r) {
      int qp = qb * 64 + wave * 16 + r + ((lane >> 4) << 3);
      localv[base + (size_t)qp * 64 + c * 16 + (lane & 15)] =
          acc[c][r] / lrow[r];
    }
}

// ---------------------------------------------------------------------------
// K3: kNN memory attention (HBM-bound 536 MB stream) + sigmoid-gate combine.
// One wave per query; phase A lanes = 32 memories, phase B lanes = d.
// ---------------------------------------------------------------------------
__global__ __launch_bounds__(256) void k_mem_attn(
    const _Float16* __restrict__ qf, const float* __restrict__ kmem,
    const float* __restrict__ vmem, const unsigned char* __restrict__ mmask,
    const float* __restrict__ nullk, const float* __restrict__ nullv,
    const float* __restrict__ gate, const float* __restrict__ localv,
    _Float16* __restrict__ comb) {
  __shared__ float qs[8][64];
  __shared__ float ps[8][32];
  const int lane = threadIdx.x & 31, wave = threadIdx.x >> 5;
  const int gq = blockIdx.x * 8 + wave;  // flat (b*8+h)*2048 + i
  const int bh = gq >> 11, h = bh & 7, b = bh >> 3;
  const int i = gq & 2047;
  const size_t qoff = (size_t)gq * 64;
  const size_t mbase = (size_t)gq * 32;

  qs[wave][lane]      = (float)qf[qoff + lane];
  qs[wave][lane + 32] = (float)qf[qoff + lane + 32];

  const float* vp = vmem + mbase * 64;
#pragma unroll
  for (int j = 0; j < 32; j += 4)  // pull v_mem rows toward L2 early
    __builtin_prefetch(vp + j * 64 + lane, 0, 1);

  // phase A: lane j = memory slot j
  const float* kp = kmem + (mbase + lane) * 64;
  float sim = 0.0f;
#pragma unroll
  for (int d4 = 0; d4 < 64; d4 += 4) {
    float4 kv = *reinterpret_cast<const float4*>(kp + d4);
    sim += kv.x * qs[wave][d4] + kv.y * qs[wave][d4 + 1] +
           kv.z * qs[wave][d4 + 2] + kv.w * qs[wave][d4 + 3];
  }
  sim *= 0.125f;
  if (!mmask[mbase + lane]) sim = MASKVAL;

  float sn = 0.0f;  // null key (uniform across lanes)
#pragma unroll
  for (int d = 0; d < 64; ++d) sn += qs[wave][d] * nullk[h * 64 + d];
  sn *= 0.125f;

  float mx = sim;
#pragma unroll
  for (int s = 16; s >= 1; s >>= 1) mx = fmaxf(mx, __shfl_xor(mx, s, 32));
  mx = fmaxf(mx, sn);
  float p = expf(sim - mx), pn = expf(sn - mx);
  float sm = p;
#pragma unroll
  for (int s = 16; s >= 1; s >>= 1) sm += __shfl_xor(sm, s, 32);
  float invd = 1.0f / (sm + pn);
  ps[wave][lane] = p;

  // phase B: lane = feature dim (and dim+32), coalesced v_mem columns
  float a0 = pn * nullv[h * 64 + lane];
  float a1 = pn * nullv[h * 64 + lane + 32];
#pragma unroll 4
  for (int j = 0; j < 32; ++j) {
    float w = ps[wave][j];
    a0 += w * vp[j * 64 + lane];
    a1 += w * vp[j * 64 + lane + 32];
  }
  float g = 1.0f / (1.0f + expf(-gate[h]));
  float o0 = g * localv[qoff + lane]      + (1.0f - g) * a0 * invd;
  float o1 = g * localv[qoff + lane + 32] + (1.0f - g) * a1 * invd;
  size_t crow = ((size_t)b * 2048 + i) * 512 + h * 64;
  comb[crow + lane]      = (_Float16)o0;
  comb[crow + lane + 32] = (_Float16)o1;
}

// ---------------------------------------------------------------------------
// K4: output projection. out(4096x512) = combined_f16 @ Wout + bout, fp32 out.
// ---------------------------------------------------------------------------
__global__ __launch_bounds__(256) void k_out_gemm(
    const _Float16* __restrict__ A, const float* __restrict__ Wout,
    const float* __restrict__ bout, float* __restrict__ out) {
  __shared__ _Float16 As[64 * 32];
  __shared__ _Float16 Bst[128 * 32];  // B transposed [N][K]
  const int tid = threadIdx.x, lane = tid & 31, wave = tid >> 5;
  const int m0 = blockIdx.y * 64, n0 = blockIdx.x * 128;
  const int wm = wave >> 2, wn = wave & 3;

  v8f acc[2][2] = {};
  for (int kk = 0; kk < 512; kk += 32) {
    __syncthreads();
    {  // stage A (already f16): 16-byte copy
      int r = tid >> 2, c = (tid & 3) * 8;
      *reinterpret_cast<float4*>(&As[r * 32 + c]) =
          *reinterpret_cast<const float4*>(A + (size_t)(m0 + r) * 512 + kk + c);
    }
    stage_wtile_T(Wout + (size_t)kk * 512 + n0, 512, Bst, tid);
    __syncthreads();
    v16h a0 = load_a_frag(As, wm * 32, 32, lane, 0);
    v16h a1 = load_a_frag(As, wm * 32 + 16, 32, lane, 0);
    v16h b0 = load_b_fragT(Bst, wn * 32, 32, lane, 0);
    v16h b1 = load_b_fragT(Bst, wn * 32 + 16, 32, lane, 0);
    acc[0][0] = wmma_f16(a0, b0, acc[0][0]);
    acc[0][1] = wmma_f16(a0, b1, acc[0][1]);
    acc[1][0] = wmma_f16(a1, b0, acc[1][0]);
    acc[1][1] = wmma_f16(a1, b1, acc[1][1]);
  }
#pragma unroll
  for (int i = 0; i < 2; ++i)
#pragma unroll
    for (int j = 0; j < 2; ++j)
#pragma unroll
      for (int r = 0; r < 8; ++r) {
        int M = m0 + wm * 32 + i * 16 + r + ((lane >> 4) << 3);
        int N = n0 + wn * 32 + j * 16 + (lane & 15);
        out[(size_t)M * 512 + N] = acc[i][j][r] + bout[N];
      }
}

// ---------------------------------------------------------------------------
extern "C" void kernel_launch(void* const* d_in, const int* in_sizes, int n_in,
                              void* d_out, int out_size, void* d_ws,
                              size_t ws_size, hipStream_t stream) {
  (void)in_sizes; (void)n_in; (void)out_size; (void)ws_size;
  const float* x     = (const float*)d_in[0];
  const float* kmem  = (const float*)d_in[1];
  const float* vmem  = (const float*)d_in[2];
  const unsigned char* mmask = (const unsigned char*)d_in[3];
  const float* Wq    = (const float*)d_in[4];
  const float* Wkv   = (const float*)d_in[5];
  const float* Wout  = (const float*)d_in[6];
  const float* bout  = (const float*)d_in[7];
  const float* rel   = (const float*)d_in[8];
  const float* nullk = (const float*)d_in[9];
  const float* nullv = (const float*)d_in[10];
  const float* gate  = (const float*)d_in[11];

  char* ws = (char*)d_ws;                       // 24 MB total
  _Float16* qf = (_Float16*)(ws);               //  4 MB q  f16 [b][h][n][64]
  _Float16* kf = (_Float16*)(ws + (4u << 20));  //  4 MB k  f16
  _Float16* vf = (_Float16*)(ws + (8u << 20));  //  4 MB v  f16
  float*    lv = (float*)   (ws + (12u << 20)); //  8 MB local_values f32
  _Float16* cb = (_Float16*)(ws + (20u << 20)); //  4 MB combined f16 [b*n][512]

  k_qkv_gemm<<<dim3(12, 64), 256, 0, stream>>>(x, Wq, Wkv, qf, kf, vf);
  k_local_attn<<<dim3(32, 16), 128, 0, stream>>>(qf, kf, vf, rel, lv);
  k_mem_attn<<<dim3(4096), 256, 0, stream>>>(qf, kmem, vmem, mmask, nullk,
                                             nullv, gate, lv, cb);
  k_out_gemm<<<dim3(4, 64), 256, 0, stream>>>(cb, Wout, bout, (float*)d_out);
}
